// Qwen3GatedDeltaNet_16836271800779
// MI455X (gfx1250) — compile-verified, matrix operand order checked by reading, not run on hardware
//
#include <hip/hip_runtime.h>
#include <hip/hip_bf16.h>
#include <stdint.h>

// ---------------- problem constants ----------------
#define B_        4
#define L_        1024
#define HID_      2048
#define H_        16
#define DK_       128
#define DV_       128
#define KS_       4
#define KEY_DIM_  2048
#define VAL_DIM_  2048
#define CONV_DIM_ 6144            // 2*KEY + VAL
#define QKVZ_N_   8192            // 2*KEY + 2*VAL
#define MROWS_    (B_ * L_)       // 4096

typedef __attribute__((ext_vector_type(16))) __bf16 v16bf;
typedef __attribute__((ext_vector_type(8)))  float  v8f;
typedef __attribute__((ext_vector_type(4)))  unsigned int u32x4;
typedef __attribute__((ext_vector_type(8)))  int i32x8;
typedef __attribute__((ext_vector_type(4)))  int i32x4;

#if defined(__gfx1250__) && __has_builtin(__builtin_amdgcn_tensor_load_to_lds) && \
    __has_builtin(__builtin_amdgcn_s_wait_tensorcnt)
#define HAVE_TDM 1
#else
#define HAVE_TDM 0
#endif

__device__ __forceinline__ unsigned short f32_to_bf16(float f) {
  unsigned int u = __float_as_uint(f);
  u += 0x7FFFu + ((u >> 16) & 1u);          // round-to-nearest-even
  return (unsigned short)(u >> 16);
}

// ---------------- fp32 -> bf16 conversion (row-major copy) ----------------
__global__ __launch_bounds__(256) void cvt_bf16_kernel(
    const float* __restrict__ src, unsigned short* __restrict__ dst, long n) {
  long i = (long)blockIdx.x * 256 + threadIdx.x;
  if (i < n) dst[i] = f32_to_bf16(src[i]);
}

// ------- fp32 (KxN) -> bf16 transposed (NxK), 32x32 LDS tile transpose -------
__global__ __launch_bounds__(256) void cvt_transpose_bf16_kernel(
    const float* __restrict__ src,      // K x N row-major
    unsigned short* __restrict__ dst,   // N x K row-major
    int K, int N) {
  __shared__ float tile[32][33];        // +1 pad: no bank conflicts
  const int tx = threadIdx.x & 31;
  const int ty = threadIdx.x >> 5;      // 0..7
  const int n0 = blockIdx.x * 32;
  const int k0 = blockIdx.y * 32;
#pragma unroll
  for (int i = 0; i < 4; ++i)
    tile[ty + 8 * i][tx] = src[(size_t)(k0 + ty + 8 * i) * N + n0 + tx];
  __syncthreads();
#pragma unroll
  for (int i = 0; i < 4; ++i)
    dst[(size_t)(n0 + ty + 8 * i) * K + k0 + tx] = f32_to_bf16(tile[tx][ty + 8 * i]);
}

#if HAVE_TDM
// Issue a 2-D TDM tile load (bf16 elements) global -> LDS, per ISA ch.8 D# layout.
// lds_addr: byte offset in LDS; gaddr: tile start in global memory.
// tile_w elements per row, tile_h rows, row stride stride0 (elements).
// 6-arg builtin form (clang-23 / therock): (g0, g1, g2, g3, g_extra, cpol).
__device__ __forceinline__ void tdm_load_2d(
    unsigned lds_addr, const void* gaddr,
    unsigned tile_w, unsigned tile_h,
    unsigned tensor_w, unsigned tensor_h, unsigned long stride0)
{
  unsigned long ga = (unsigned long)gaddr;
  u32x4 g0;
  g0[0] = 1u;                                            // count=1, user mode
  g0[1] = lds_addr;                                      // lds_addr [63:32]
  g0[2] = (unsigned)(ga & 0xFFFFFFFFu);                  // global_addr lo
  g0[3] = (unsigned)((ga >> 32) & 0x01FFFFFFu) | (2u << 30);  // addr[56:32] | type=2
  i32x8 g1;
  g1[0] = (int)(1u << 16);                               // data_size=1 (2B), mask=0
  g1[1] = (int)((tensor_w & 0xFFFFu) << 16);             // tensor_dim0[15:0]
  g1[2] = (int)(((tensor_w >> 16) & 0xFFFFu) | ((tensor_h & 0xFFFFu) << 16));
  g1[3] = (int)(((tensor_h >> 16) & 0xFFFFu) | ((tile_w & 0xFFFFu) << 16));
  g1[4] = (int)(tile_h & 0xFFFFu);                       // tile_dim1 (tile_dim2=0)
  g1[5] = (int)(unsigned)(stride0 & 0xFFFFFFFFu);        // tensor_dim0_stride lo
  g1[6] = (int)(unsigned)((stride0 >> 32) & 0xFFFFu);    // stride hi16 (dim1_stride=0)
  g1[7] = 0;
  i32x4 z4 = {0, 0, 0, 0};
  i32x8 z8 = {0, 0, 0, 0, 0, 0, 0, 0};
  __builtin_amdgcn_tensor_load_to_lds(g0, g1, z4, z4, z8, 0);
}
#endif

// ------- bf16 WMMA GEMM: C(f32, MxN) = A(bf16, MxK) * Bt(bf16, NxK)^T --------
// Bt is the transposed operand (N x K row-major), so both A and B tiles are
// (rows x 32) in LDS and both fragment loads collapse to 2x ds_load_b128.
// M % 128 == 0, N % 256 == 0, K % 32 == 0. Block = 256 threads = 8 waves.
// Block tile 128x256, K-step 32, double-buffered LDS fed by TDM.
#define BM 128
#define BN 256
#define BK 32

union FragBF { v16bf v; unsigned short h[16]; unsigned int w[8]; };

__global__ __launch_bounds__(256) void gemm_bf16_wmma_kernel(
    const unsigned short* __restrict__ A,    // M x K row-major
    const unsigned short* __restrict__ Bt,   // N x K row-major (B transposed)
    float* __restrict__ C,                   // M x N row-major
    int M, int N, int K)
{
  __shared__ __align__(16) unsigned short As[2][BM * BK];   // 2 x 8 KB
  __shared__ __align__(16) unsigned short Bs[2][BN * BK];   // 2 x 16 KB

  const int tid   = threadIdx.x;
  const int lane  = tid & 31;
  const int wave  = tid >> 5;        // 0..7
  const int wrow  = wave >> 2;       // 0..1  (64 rows each)
  const int wcol  = wave & 3;        // 0..3  (64 cols each)
  const int l16   = lane & 15;
  const int lhalf = lane >> 4;       // 0/1 -> K offset +8 (ISA 16-bit layouts)

  const int bm0 = blockIdx.y * BM;
  const int bn0 = blockIdx.x * BN;
  const int nk  = K / BK;

  v8f acc[4][4] = {};

#if HAVE_TDM
  if (wave == 0) {      // prologue: stage k-tile 0 into buffer 0
    tdm_load_2d((unsigned)(uintptr_t)&As[0][0], A + (size_t)bm0 * K,
                BK, BM, (unsigned)K, (unsigned)(M - bm0), (unsigned long)K);
    tdm_load_2d((unsigned)(uintptr_t)&Bs[0][0], Bt + (size_t)bn0 * K,
                BK, BN, (unsigned)K, (unsigned)(N - bn0), (unsigned long)K);
  }
#endif

  for (int kt = 0; kt < nk; ++kt) {
    const int cur = kt & 1;
#if HAVE_TDM
    if (wave == 0) {
      if (kt + 1 < nk) {               // stage next tile into the other buffer
        int k1 = (kt + 1) * BK;
        tdm_load_2d((unsigned)(uintptr_t)&As[cur ^ 1][0],
                    A + (size_t)bm0 * K + k1,
                    BK, BM, (unsigned)(K - k1), (unsigned)(M - bm0),
                    (unsigned long)K);
        tdm_load_2d((unsigned)(uintptr_t)&Bs[cur ^ 1][0],
                    Bt + (size_t)bn0 * K + k1,
                    BK, BN, (unsigned)(K - k1), (unsigned)(N - bn0),
                    (unsigned long)K);
        __builtin_amdgcn_s_wait_tensorcnt(2);   // current 2 tiles complete
      } else {
        __builtin_amdgcn_s_wait_tensorcnt(0);
      }
    }
    __syncthreads();
#else
    {   // fallback: cooperative vector loads
      int r = tid >> 1, c = (tid & 1) << 4;
      const uint4* s = (const uint4*)(A + (size_t)(bm0 + r) * K + kt * BK + c);
      uint4* d = (uint4*)(&As[cur][r * BK + c]);
      d[0] = s[0]; d[1] = s[1];
      const uint4* sb = (const uint4*)(Bt + (size_t)(bn0 + tid) * K + kt * BK);
      uint4* db = (uint4*)(&Bs[cur][tid * BK]);
      db[0] = sb[0]; db[1] = sb[1]; db[2] = sb[2]; db[3] = sb[3];
    }
    __syncthreads();
#endif

    // fragments (ISA §7.12.2): lanes 0-15 -> K {0..7,16..23}, lanes 16-31 -> +8;
    // both runs are contiguous 16B in a (row x 32) LDS tile -> 2x ds_load_b128.
    FragBF a[4];
#pragma unroll
    for (int i = 0; i < 4; ++i) {
      int m = wrow * 64 + i * 16 + l16;
#pragma unroll
      for (int p = 0; p < 8; ++p) {
        int kp = (p < 4 ? 2 * p : 2 * p + 8) + 8 * lhalf;
        a[i].w[p] = *(const unsigned int*)(&As[cur][m * BK + kp]);
      }
    }
#pragma unroll
    for (int j = 0; j < 4; ++j) {
      FragBF b;
      int n = wcol * 64 + j * 16 + l16;
#pragma unroll
      for (int p = 0; p < 8; ++p) {
        int kp = (p < 4 ? 2 * p : 2 * p + 8) + 8 * lhalf;
        b.w[p] = *(const unsigned int*)(&Bs[cur][n * BK + kp]);
      }
#pragma unroll
      for (int i = 0; i < 4; ++i)
        acc[i][j] = __builtin_amdgcn_wmma_f32_16x16x32_bf16(
            false, a[i].v, false, b.v, (short)0, acc[i][j], false, false);
    }
    __syncthreads();   // everyone done reading before buffer is overwritten
  }

  // C/D layout: VGPR r -> row r (lanes 0-15) / r+8 (lanes 16-31), col = l16
#pragma unroll
  for (int i = 0; i < 4; ++i) {
#pragma unroll
    for (int j = 0; j < 4; ++j) {
      int col = bn0 + wcol * 64 + j * 16 + l16;
      float* p = C + (size_t)(bm0 + wrow * 64 + i * 16 + 8 * lhalf) * N + col;
#pragma unroll
      for (int r = 0; r < 8; ++r) p[(size_t)r * N] = acc[i][j][r];
    }
  }
}

// ---------------- tiny GEMM for ba: (4096 x 2048) @ (2048 x 32), f32 ---------
__global__ __launch_bounds__(256) void ba_gemm_kernel(
    const float* __restrict__ hs, const float* __restrict__ Wba,
    float* __restrict__ ba) {
  long i = (long)blockIdx.x * 256 + threadIdx.x;   // 4096*32
  if (i >= (long)MROWS_ * 32) return;
  int  n = (int)(i & 31);
  long m = i >> 5;
  const float* a = hs + m * HID_;
  float acc = 0.f;
  for (int k = 0; k < HID_; ++k) acc = fmaf(a[k], Wba[(long)k * 32 + n], acc);
  ba[i] = acc;
}

// ---------------- causal depthwise conv (KS=4) + bias -----------------------
__global__ __launch_bounds__(256) void conv_kernel(
    const float* __restrict__ qkvz, const float* __restrict__ conv_w,
    const float* __restrict__ conv_b, float* __restrict__ x) {
  long i = (long)blockIdx.x * 256 + threadIdx.x;   // B*L*CONV_DIM
  if (i >= (long)MROWS_ * CONV_DIM_) return;
  int  c  = (int)(i % CONV_DIM_);
  long bt = i / CONV_DIM_;
  int  t  = (int)(bt % L_);
  long b  = bt / L_;
  float acc = conv_b[c];
#pragma unroll
  for (int j = 0; j < KS_; ++j) {
    int tt = t - (KS_ - 1) + j;
    if (tt >= 0)
      acc = fmaf(qkvz[(b * L_ + tt) * (long)QKVZ_N_ + c], conv_w[c * KS_ + j], acc);
  }
  x[i] = acc;
}

// ---------------- l2norm over contiguous 128-rows (q and k halves) ----------
__global__ __launch_bounds__(256) void l2norm_kernel(float* __restrict__ x) {
  long gid = (long)blockIdx.x * 256 + threadIdx.x;
  long wid = gid >> 5;                 // one wave per 128-wide row; rows = B*L*32
  int lane = threadIdx.x & 31;
  long bt  = wid >> 5;
  int  rin = (int)(wid & 31);
  float* p = x + bt * (long)CONV_DIM_ + (long)rin * 128 + lane * 4;
  float4 v = *(float4*)p;
  float ss = v.x * v.x + v.y * v.y + v.z * v.z + v.w * v.w;
#pragma unroll
  for (int m = 16; m >= 1; m >>= 1) ss += __shfl_xor(ss, m, 32);
  float r = rsqrtf(ss + 1e-6f);
  v.x *= r; v.y *= r; v.z *= r; v.w *= r;
  *(float4*)p = v;
}

// ---------------- g = -exp(A_log)*softplus(a+dt_bias); beta = sigmoid(b) -----
__global__ __launch_bounds__(256) void gbeta_kernel(
    const float* __restrict__ ba, const float* __restrict__ A_log,
    const float* __restrict__ dt_bias, float* __restrict__ g,
    float* __restrict__ beta) {
  long i = (long)blockIdx.x * 256 + threadIdx.x;   // B*L*H
  if (i >= (long)MROWS_ * H_) return;
  int  h  = (int)(i & 15);
  long bt = i >> 4;
  float bv = ba[bt * 32 + h];
  float av = ba[bt * 32 + 16 + h];
  float sx = av + dt_bias[h];
  float sp = sx > 20.f ? sx : log1pf(__expf(sx));
  g[i]    = -__expf(A_log[h]) * sp;
  beta[i] = 1.f / (1.f + __expf(-bv));
}

// ---------------- gated delta-rule scan: one block per (b,h) -----------------
// thread d owns state column S[:,d] (128 VGPRs); k_t/q_t broadcast via LDS.
__global__ __launch_bounds__(128) void gdn_scan_kernel(
    const float* __restrict__ x,      // (B,L,CONV_DIM): q|k|v
    const float* __restrict__ g,      // (B,L,H)
    const float* __restrict__ beta,   // (B,L,H)
    float* __restrict__ o)            // (B,L,H,DV)
{
  const int bh = blockIdx.x;          // 0..63
  const int b  = bh >> 4;
  const int h  = bh & 15;
  const int d  = threadIdx.x;         // 0..127
  const float scale = 0.08838834764831845f;   // 128^-0.5

  __shared__ __align__(16) float kq[2 * DK_];   // [0,128)=k_t, [128,256)=q_t

  float S[DK_];
#pragma unroll
  for (int i = 0; i < DK_; ++i) S[i] = 0.f;

  const long baseBL = (long)b * L_;
  const float4* k4 = (const float4*)kq;
  const float4* q4 = (const float4*)(kq + DK_);

  for (int t = 0; t < L_; ++t) {
    const long row = (baseBL + t) * (long)CONV_DIM_;
    float qv = x[row + (long)h * DK_ + d];
    float kv = x[row + KEY_DIM_ + (long)h * DK_ + d];
    float vv = x[row + 2 * KEY_DIM_ + (long)h * DV_ + d];
    __syncthreads();                  // previous step's readers done
    kq[d]       = kv;
    kq[DK_ + d] = qv;
    __syncthreads();

    float gt  = g[(baseBL + t) * H_ + h];
    float bt_ = beta[(baseBL + t) * H_ + h];
    float eg  = __expf(gt);

    float vp = 0.f;
#pragma unroll
    for (int c = 0; c < DK_ / 4; ++c) {          // S *= exp(g); vp = k . S
      float4 kvv = k4[c];
      float s0 = S[4 * c + 0] * eg; vp = fmaf(kvv.x, s0, vp); S[4 * c + 0] = s0;
      float s1 = S[4 * c + 1] * eg; vp = fmaf(kvv.y, s1, vp); S[4 * c + 1] = s1;
      float s2 = S[4 * c + 2] * eg; vp = fmaf(kvv.z, s2, vp); S[4 * c + 2] = s2;
      float s3 = S[4 * c + 3] * eg; vp = fmaf(kvv.w, s3, vp); S[4 * c + 3] = s3;
    }
    float delta = (vv - vp) * bt_;
    float ov = 0.f;
#pragma unroll
    for (int c = 0; c < DK_ / 4; ++c) {          // S += outer(k,delta); o = q . S
      float4 kvv = k4[c];
      float4 qvv = q4[c];
      float s0 = fmaf(kvv.x, delta, S[4 * c + 0]); S[4 * c + 0] = s0; ov = fmaf(qvv.x, s0, ov);
      float s1 = fmaf(kvv.y, delta, S[4 * c + 1]); S[4 * c + 1] = s1; ov = fmaf(qvv.y, s1, ov);
      float s2 = fmaf(kvv.z, delta, S[4 * c + 2]); S[4 * c + 2] = s2; ov = fmaf(qvv.z, s2, ov);
      float s3 = fmaf(kvv.w, delta, S[4 * c + 3]); S[4 * c + 3] = s3; ov = fmaf(qvv.w, s3, ov);
    }
    o[((baseBL + t) * H_ + h) * (long)DV_ + d] = ov * scale;
  }
}

// ---------------- RMS norm * norm_w * sigmoid(z), emit bf16 ------------------
__global__ __launch_bounds__(256) void gate_norm_kernel(
    const float* __restrict__ o, const float* __restrict__ qkvz,
    const float* __restrict__ norm_w, unsigned short* __restrict__ og) {
  long gid = (long)blockIdx.x * 256 + threadIdx.x;
  long wid = gid >> 5;                // row = (b*L+t)*H + h ; rows = 65536
  int lane = threadIdx.x & 31;
  long bt  = wid >> 4;
  int  h   = (int)(wid & 15);
  const float* orow = o + wid * DV_;
  float4 ov = *(const float4*)(orow + lane * 4);
  float ss = ov.x * ov.x + ov.y * ov.y + ov.z * ov.z + ov.w * ov.w;
#pragma unroll
  for (int m = 16; m >= 1; m >>= 1) ss += __shfl_xor(ss, m, 32);
  float r = rsqrtf(ss * (1.0f / DV_) + 1e-6f);
  float4 zv = *(const float4*)(qkvz + bt * (long)QKVZ_N_ + CONV_DIM_ + (long)h * DV_ + lane * 4);
  float4 nw = *(const float4*)(norm_w + lane * 4);
  unsigned short* dst = og + bt * (long)VAL_DIM_ + (long)h * DV_ + lane * 4;
  dst[0] = f32_to_bf16(ov.x * r * nw.x * (1.f / (1.f + __expf(-zv.x))));
  dst[1] = f32_to_bf16(ov.y * r * nw.y * (1.f / (1.f + __expf(-zv.y))));
  dst[2] = f32_to_bf16(ov.z * r * nw.z * (1.f / (1.f + __expf(-zv.z))));
  dst[3] = f32_to_bf16(ov.w * r * nw.w * (1.f / (1.f + __expf(-zv.w))));
}

// ---------------- launcher ----------------
extern "C" void kernel_launch(void* const* d_in, const int* in_sizes, int n_in,
                              void* d_out, int out_size, void* d_ws, size_t ws_size,
                              hipStream_t stream) {
  const float* hs      = (const float*)d_in[0];
  const float* Wqkvz   = (const float*)d_in[1];
  const float* Wba     = (const float*)d_in[2];
  const float* conv_w  = (const float*)d_in[3];
  const float* conv_b  = (const float*)d_in[4];
  const float* A_log   = (const float*)d_in[5];
  const float* dt_bias = (const float*)d_in[6];
  const float* norm_w  = (const float*)d_in[7];
  const float* Wout    = (const float*)d_in[8];
  float* out = (float*)d_out;

  char* ws = (char*)d_ws;
  size_t off = 0;
  auto alloc = [&](size_t bytes) -> void* {
    void* p = ws + off;
    off = (off + bytes + 255) & ~(size_t)255;
    return p;
  };

  unsigned short* hs_bf    = (unsigned short*)alloc((size_t)MROWS_ * HID_ * 2);
  unsigned short* WqkvzT   = (unsigned short*)alloc((size_t)HID_ * QKVZ_N_ * 2);  // N x K
  unsigned short* WoutT    = (unsigned short*)alloc((size_t)VAL_DIM_ * HID_ * 2); // N x K
  float*          qkvz     = (float*)alloc((size_t)MROWS_ * QKVZ_N_ * 4);
  float*          ba       = (float*)alloc((size_t)MROWS_ * 32 * 4);
  float*          xconv    = (float*)alloc((size_t)MROWS_ * CONV_DIM_ * 4);
  float*          gbuf     = (float*)alloc((size_t)MROWS_ * H_ * 4);
  float*          bbuf     = (float*)alloc((size_t)MROWS_ * H_ * 4);
  float*          obuf     = (float*)alloc((size_t)MROWS_ * VAL_DIM_ * 4);
  unsigned short* og_bf    = (unsigned short*)alloc((size_t)MROWS_ * VAL_DIM_ * 2);

  // 1) fp32 -> bf16 (A row-major; weights transposed to N x K)
  {
    long n = (long)MROWS_ * HID_;
    cvt_bf16_kernel<<<(int)((n + 255) / 256), 256, 0, stream>>>(hs, hs_bf, n);
  }
  {
    dim3 grid(QKVZ_N_ / 32, HID_ / 32);
    cvt_transpose_bf16_kernel<<<grid, 256, 0, stream>>>(Wqkvz, WqkvzT, HID_, QKVZ_N_);
  }
  {
    dim3 grid(HID_ / 32, VAL_DIM_ / 32);
    cvt_transpose_bf16_kernel<<<grid, 256, 0, stream>>>(Wout, WoutT, VAL_DIM_, HID_);
  }

  // 2) qkvz = hs @ W_qkvz   (4096 x 2048) * (2048 x 8192)
  {
    dim3 grid(QKVZ_N_ / BN, MROWS_ / BM);
    gemm_bf16_wmma_kernel<<<grid, 256, 0, stream>>>(hs_bf, WqkvzT, qkvz,
                                                    MROWS_, QKVZ_N_, HID_);
  }

  // 3) ba = hs @ W_ba (N=32, plain f32)
  {
    long n = (long)MROWS_ * 32;
    ba_gemm_kernel<<<(int)((n + 255) / 256), 256, 0, stream>>>(hs, Wba, ba);
  }

  // 4) causal depthwise conv + bias over first CONV_DIM channels
  {
    long n = (long)MROWS_ * CONV_DIM_;
    conv_kernel<<<(int)((n + 255) / 256), 256, 0, stream>>>(qkvz, conv_w, conv_b, xconv);
  }

  // 5) l2norm of q and k (first 4096 channels, rows of 128)
  {
    long waves = (long)MROWS_ * 32;
    l2norm_kernel<<<(int)(waves / 8), 256, 0, stream>>>(xconv);
  }

  // 6) g / beta
  {
    long n = (long)MROWS_ * H_;
    gbeta_kernel<<<(int)((n + 255) / 256), 256, 0, stream>>>(ba, A_log, dt_bias, gbuf, bbuf);
  }

  // 7) gated delta-rule scan
  gdn_scan_kernel<<<B_ * H_, 128, 0, stream>>>(xconv, gbuf, bbuf, obuf);

  // 8) RMS-norm * gate -> bf16
  {
    long waves = (long)MROWS_ * H_;
    gate_norm_kernel<<<(int)(waves / 8), 256, 0, stream>>>(obuf, qkvz, norm_w, og_bf);
  }

  // 9) out = og @ W_out   (4096 x 2048) * (2048 x 2048)
  {
    dim3 grid(HID_ / BN, MROWS_ / BM);
    gemm_bf16_wmma_kernel<<<grid, 256, 0, stream>>>(og_bf, WoutT, out,
                                                    MROWS_, HID_, VAL_DIM_);
  }
}
